// LocalDecoder_51780125721276
// MI455X (gfx1250) — matrix operand order repaired and checked by reading, not compiled, not run on hardware
//
#include <hip/hip_runtime.h>
#include <hip/hip_bf16.h>

typedef __attribute__((ext_vector_type(16))) __bf16        v16bf;
typedef __attribute__((ext_vector_type(8)))  float         v8f;
typedef __attribute__((ext_vector_type(8)))  unsigned int  v8u;
typedef __attribute__((ext_vector_type(4)))  int           v4i;

// ---- async global->LDS path (gfx1250), guarded so sync fallback compiles ----
#if defined(__has_builtin)
#  if __has_builtin(__builtin_amdgcn_global_load_async_to_lds_b128) && \
      __has_builtin(__builtin_amdgcn_s_wait_asynccnt)
#    define USE_ASYNC_LDS 1
#  else
#    define USE_ASYNC_LDS 0
#  endif
#else
#  define USE_ASYNC_LDS 0
#endif

// ---------- bf16 helpers ----------
__device__ __forceinline__ unsigned short f2bf(float f) {
  unsigned int u = __builtin_bit_cast(unsigned int, f);
  unsigned int r = u + 0x7FFFu + ((u >> 16) & 1u);   // round-to-nearest-even
  return (unsigned short)(r >> 16);
}
__device__ __forceinline__ float bf2f(unsigned int h) {
  unsigned int u = h << 16;
  return __builtin_bit_cast(float, u);
}
__device__ __forceinline__ int iclamp(int v, int lo, int hi) {
  return v < lo ? lo : (v > hi ? hi : v);
}

// ---------- bilinear sampler state ----------
struct Samp {
  float w00, w10, w01, w11;
  int   o00, o10, o01, o11;
};

__device__ __forceinline__ Samp make_samp(float un, float vn, int S, float valid) {
  float x  = ((un + 1.f) * (float)S - 1.f) * 0.5f;
  float y  = ((vn + 1.f) * (float)S - 1.f) * 0.5f;
  float x0 = floorf(x), y0 = floorf(y);
  float wx1 = x - x0, wx0 = 1.f - wx1;
  float wy1 = y - y0, wy0 = 1.f - wy1;
  int x0i = (int)x0, y0i = (int)y0;
  int x1i = x0i + 1, y1i = y0i + 1;
  float vx0 = (x0i >= 0 && x0i < S) ? 1.f : 0.f;
  float vx1 = (x1i >= 0 && x1i < S) ? 1.f : 0.f;
  float vy0 = (y0i >= 0 && y0i < S) ? 1.f : 0.f;
  float vy1 = (y1i >= 0 && y1i < S) ? 1.f : 0.f;
  int x0c = iclamp(x0i, 0, S - 1), x1c = iclamp(x1i, 0, S - 1);
  int y0c = iclamp(y0i, 0, S - 1), y1c = iclamp(y1i, 0, S - 1);
  Samp s;
  s.w00 = wx0 * wy0 * vx0 * vy0 * valid;
  s.w10 = wx1 * wy0 * vx1 * vy0 * valid;
  s.w01 = wx0 * wy1 * vx0 * vy1 * valid;
  s.w11 = wx1 * wy1 * vx1 * vy1 * valid;
  s.o00 = y0c * S + x0c;  s.o10 = y0c * S + x1c;
  s.o01 = y1c * S + x0c;  s.o11 = y1c * S + x1c;
  return s;
}

__device__ __forceinline__ float sample_ch(const float* __restrict__ g, const Samp& s, int plane) {
  return s.w00 * g[plane + s.o00] + s.w10 * g[plane + s.o10] +
         s.w01 * g[plane + s.o01] + s.w11 * g[plane + s.o11];
}

// =====================================================================
// Kernel 0: pack weights to bf16, transposed to [K][N] row-major
// =====================================================================
__global__ void prep_weights(const float* __restrict__ in_w,
                             const float* __restrict__ blk_w1,
                             const float* __restrict__ blk_w2,
                             unsigned short* __restrict__ Win,
                             unsigned short* __restrict__ W1,
                             unsigned short* __restrict__ W2) {
  const int NIN = 1280 * 512;
  const int NB  = 512 * 512;
  int idx = blockIdx.x * blockDim.x + threadIdx.x;
  if (idx < NIN) {
    int k = idx / 512, o = idx % 512;
    Win[idx] = f2bf(in_w[o * 1280 + k]);
  } else if (idx < NIN + 4 * NB) {
    int r = idx - NIN;
    int blk = r / NB, j = r % NB;
    int k = j / 512, o = j % 512;
    W1[r] = f2bf(blk_w1[blk * NB + o * 512 + k]);
  } else if (idx < NIN + 8 * NB) {
    int r = idx - NIN - 4 * NB;
    int blk = r / NB, j = r % NB;
    int k = j / 512, o = j % 512;
    W2[r] = f2bf(blk_w2[blk * NB + o * 512 + k]);
  }
}

// =====================================================================
// Kernel 1: featurize — projection, bilinear gathers, small MLPs, gate.
// Writes comb[M=131072][1280] in bf16 row-major.
// =====================================================================
__global__ __launch_bounds__(256)
void featurize(const float* __restrict__ g0, const float* __restrict__ g1,
               const float* __restrict__ g2, const float* __restrict__ g3,
               const float* __restrict__ hf,
               const float* __restrict__ points, const float* __restrict__ gfeat,
               const float* __restrict__ scale, const float* __restrict__ loc,
               const float* __restrict__ Kc, const float* __restrict__ RT,
               const float* __restrict__ pt_w1, const float* __restrict__ pt_b1,
               const float* __restrict__ pt_w2, const float* __restrict__ pt_b2,
               const float* __restrict__ z_w1, const float* __restrict__ z_b1,
               const float* __restrict__ z_w2, const float* __restrict__ z_b2,
               const float* __restrict__ g_w1, const float* __restrict__ g_b1,
               const float* __restrict__ g_w2, const float* __restrict__ g_b2,
               unsigned short* __restrict__ comb) {
  int id = blockIdx.x * blockDim.x + threadIdx.x;
  if (id >= 131072) return;
  int b = id >> 14;

  // ---- projection ----
  float sc = scale[b];
  const float* lc = loc + b * 3;
  const float* pp = points + (size_t)id * 3;
  float px = pp[0], py = pp[1], pz = pp[2];
  float X = px * sc + lc[0], Y = py * sc + lc[1], Z = pz * sc + lc[2];
  const float* rt = RT + b * 12;
  float cx = rt[0] * X + rt[1] * Y + rt[2]  * Z + rt[3];
  float cy = rt[4] * X + rt[5] * Y + rt[6]  * Z + rt[7];
  float cz = rt[8] * X + rt[9] * Y + rt[10] * Z + rt[11];
  const float* km = Kc + b * 9;
  float ix = km[0] * cx + km[1] * cy + km[2] * cz;
  float iy = km[3] * cx + km[4] * cy + km[5] * cz;
  float iz = km[6] * cx + km[7] * cy + km[8] * cz;
  float zd = iz + 1e-8f;
  float u = ix / zd, v = iy / zd;
  float valid = iz > 0.f ? 1.f : 0.f;
  float un = (2.f * u + 1.f) / 224.f - 1.f;
  float vn = (2.f * v + 1.f) / 224.f - 1.f;

  Samp s56 = make_samp(un, vn, 56, valid);
  Samp s28 = make_samp(un, vn, 28, valid);
  Samp s14 = make_samp(un, vn, 14, valid);
  Samp s7  = make_samp(un, vn, 7,  valid);

  // ---- z encoder: 1 -> 16 relu -> 16 ----
  float zh[16], ze[16];
  #pragma unroll
  for (int o = 0; o < 16; ++o) zh[o] = fmaxf(z_w1[o] * cz + z_b1[o], 0.f);
  #pragma unroll
  for (int o = 0; o < 16; ++o) {
    float a = z_b2[o];
    #pragma unroll
    for (int k = 0; k < 16; ++k) a += z_w2[o * 16 + k] * zh[k];
    ze[o] = a;
  }

  // ---- gate: 416 -> 64 relu -> 1 sigmoid ----
  float gh[64];
  #pragma unroll
  for (int o = 0; o < 64; ++o) gh[o] = g_b1[o];
  const float* gf = gfeat + b * 256;
  for (int j = 0; j < 256; ++j) {
    float x = gf[j];
    #pragma unroll
    for (int o = 0; o < 64; ++o) gh[o] += g_w1[o * 416 + j] * x;
  }
  for (int c = 0; c < 144; ++c) {
    float x = sample_ch(hf, s56, (b * 144 + c) * 3136);
    int j = 256 + c;
    #pragma unroll
    for (int o = 0; o < 64; ++o) gh[o] += g_w1[o * 416 + j] * x;
  }
  for (int c = 0; c < 16; ++c) {
    float x = ze[c];
    int j = 400 + c;
    #pragma unroll
    for (int o = 0; o < 64; ++o) gh[o] += g_w1[o * 416 + j] * x;
  }
  float gacc = g_b2[0];
  #pragma unroll
  for (int o = 0; o < 64; ++o) gacc += g_w2[o] * fmaxf(gh[o], 0.f);
  float gate2 = 2.f / (1.f + expf(-gacc));

  // ---- write comb row (bf16, paired dword stores) ----
  size_t rowbase = (size_t)id * 1280;
  unsigned short* crow = comb + rowbase;
  auto put2 = [&](int k, float a, float c2) {
    *(unsigned int*)(crow + k) = (unsigned int)f2bf(a) | ((unsigned int)f2bf(c2) << 16);
  };

  // local features, gated: grid0[0..63], grid1[64..191], grid2[192..447], grid3[448..959]
  for (int c = 0; c < 64; c += 2)
    put2(c,       sample_ch(g0, s56, (b * 64  + c) * 3136) * gate2,
                  sample_ch(g0, s56, (b * 64  + c + 1) * 3136) * gate2);
  for (int c = 0; c < 128; c += 2)
    put2(64 + c,  sample_ch(g1, s28, (b * 128 + c) * 784) * gate2,
                  sample_ch(g1, s28, (b * 128 + c + 1) * 784) * gate2);
  for (int c = 0; c < 256; c += 2)
    put2(192 + c, sample_ch(g2, s14, (b * 256 + c) * 196) * gate2,
                  sample_ch(g2, s14, (b * 256 + c + 1) * 196) * gate2);
  for (int c = 0; c < 512; c += 2)
    put2(448 + c, sample_ch(g3, s7,  (b * 512 + c) * 49) * gate2,
                  sample_ch(g3, s7,  (b * 512 + c + 1) * 49) * gate2);

  // ---- pt encoder: 3 -> 64 relu -> 64, raw points ----
  float h1[64];
  #pragma unroll
  for (int o = 0; o < 64; ++o)
    h1[o] = fmaxf(pt_w1[o * 3] * px + pt_w1[o * 3 + 1] * py + pt_w1[o * 3 + 2] * pz + pt_b1[o], 0.f);
  for (int o = 0; o < 64; o += 2) {
    float a = pt_b2[o], c2 = pt_b2[o + 1];
    #pragma unroll
    for (int k = 0; k < 64; ++k) {
      a  += pt_w2[o * 64 + k] * h1[k];
      c2 += pt_w2[(o + 1) * 64 + k] * h1[k];
    }
    put2(960 + o, a, c2);
  }

  // ---- g_exp copy ----
  for (int j = 0; j < 256; j += 2) put2(1024 + j, gf[j], gf[j + 1]);
}

// =====================================================================
// Kernel 2: bf16 WMMA GEMM.  Out[M][N] = act( A[M][K] @ Bw[K][N] + bias
//                                             (+ resid[M][N]) )
// Block = 256 threads = 8 waves; wave -> 32x64 tile (2 A frags x 4 B
// frags = 8 WMMA per 32-K step, B loaded once). B strip staged via LDS
// in 128-K chunks (async global->LDS when available).
// Requires K % 128 == 0, N % 64 == 0, M % 256 == 0.
// =====================================================================
__global__ __launch_bounds__(256)
void gemm_bf16(const unsigned short* __restrict__ A,
               const unsigned short* __restrict__ Bw,
               const float* __restrict__ bias,
               const unsigned short* __restrict__ resid,
               unsigned short* __restrict__ Out,
               int K, int N, int relu) {
  __shared__ uint4 ldsB[1024];           // 128 rows x 64 cols bf16 = 16 KB

  const int tid    = threadIdx.x;
  const int wave   = tid >> 5;
  const int lane   = tid & 31;
  const int laneHi = lane >> 4;          // 0/1
  const int laneLo = lane & 15;
  const int mBase  = (blockIdx.x * 8 + wave) * 32;
  const int colBase = blockIdx.y * 64;
  const size_t aRow0 = (size_t)(mBase + laneLo) * (size_t)K;
  const size_t aRow1 = aRow0 + (size_t)16 * (size_t)K;

  v8f acc[8];
  #pragma unroll
  for (int i = 0; i < 8; ++i) acc[i] = v8f{};

  const int chunks = K >> 7;
  for (int kc = 0; kc < chunks; ++kc) {
    __syncthreads();
    // ---- stage B chunk: 128 rows x 64 cols bf16 ----
    const unsigned short* src = Bw + (size_t)(kc * 128) * (size_t)N + colBase;
#if USE_ASYNC_LDS
    #pragma unroll
    for (int vv = 0; vv < 4; ++vv) {
      int vec = tid * 4 + vv;            // 0..1023
      int row = vec >> 3, off = vec & 7;
      const uint4* gp = ((const uint4*)(src + (size_t)row * (size_t)N)) + off;
      __builtin_amdgcn_global_load_async_to_lds_b128(
          (v4i*)gp,
          (__attribute__((address_space(3))) v4i*)&ldsB[vec], 0, 0);
    }
    __builtin_amdgcn_s_wait_asynccnt(0);
#else
    #pragma unroll
    for (int vv = 0; vv < 4; ++vv) {
      int vec = tid * 4 + vv;            // 0..1023
      int row = vec >> 3, off = vec & 7;
      ldsB[vec] = ((const uint4*)(src + (size_t)row * (size_t)N))[off];
    }
#endif
    __syncthreads();

    #pragma unroll
    for (int t4 = 0; t4 < 4; ++t4) {
      const size_t kOff = (size_t)(kc * 128 + t4 * 32 + laneHi * 8);
      // two A fragments (rows mBase..+15 and mBase+16..+31)
      const uint4* ap0 = (const uint4*)(A + aRow0 + kOff);
      const uint4* ap1 = (const uint4*)(A + aRow1 + kOff);
      uint4 a0l = ap0[0], a0h = ap0[2];  // +16 bf16 = +32 B
      uint4 a1l = ap1[0], a1h = ap1[2];
      v8u a0u = {a0l.x, a0l.y, a0l.z, a0l.w, a0h.x, a0h.y, a0h.z, a0h.w};
      v8u a1u = {a1l.x, a1l.y, a1l.z, a1l.w, a1h.x, a1h.y, a1h.z, a1h.w};
      v16bf A0 = __builtin_bit_cast(v16bf, a0u);
      v16bf A1 = __builtin_bit_cast(v16bf, a1u);

      // four B fragments, loaded once for both A fragments
      // lane L holds B column n=L%16; element s is K = 8*(L/16)+s (s<8), 16+8*(L/16)+(s-8) (s>=8)
      const unsigned short* lb = (const unsigned short*)ldsB;
      v16bf Bf[4];
      #pragma unroll
      for (int nt = 0; nt < 4; ++nt) {
        v16bf bv;
        #pragma unroll
        for (int s = 0; s < 16; ++s) {
          int kr = t4 * 32 + laneHi * 8 + (s < 8 ? s : s + 8);
          bv[s] = __builtin_bit_cast(__bf16, lb[kr * 64 + nt * 16 + laneLo]);
        }
        Bf[nt] = bv;
      }

      #pragma unroll
      for (int nt = 0; nt < 4; ++nt)
        acc[nt] = __builtin_amdgcn_wmma_f32_16x16x32_bf16(
            false, A0, false, Bf[nt], (short)0, acc[nt], false, false);
      #pragma unroll
      for (int nt = 0; nt < 4; ++nt)
        acc[4 + nt] = __builtin_amdgcn_wmma_f32_16x16x32_bf16(
            false, A1, false, Bf[nt], (short)0, acc[4 + nt], false, false);
    }
  }

  // ---- epilogue: bias + optional residual + optional relu, store bf16 ----
  #pragma unroll
  for (int half = 0; half < 2; ++half) {
    #pragma unroll
    for (int nt = 0; nt < 4; ++nt) {
      v8f a = acc[half * 4 + nt];
      int n = colBase + nt * 16 + laneLo;
      float bv = bias[n];
      #pragma unroll
      for (int i = 0; i < 8; ++i) {
        int m = mBase + half * 16 + i + laneHi * 8;
        float vr = a[i] + bv;
        if (resid) vr += bf2f(resid[(size_t)m * (size_t)N + n]);
        if (relu) vr = fmaxf(vr, 0.f);
        Out[(size_t)m * (size_t)N + n] = f2bf(vr);
      }
    }
  }
}

// =====================================================================
// Kernel 3: final 512 -> 1 projection, f32 output (vectorized loads)
// =====================================================================
__global__ __launch_bounds__(256)
void out_layer(const unsigned short* __restrict__ net,
               const float* __restrict__ w, const float* __restrict__ b,
               float* __restrict__ out, int M) {
  int m = blockIdx.x * blockDim.x + threadIdx.x;
  if (m >= M) return;
  const uint4* row = (const uint4*)(net + (size_t)m * 512);  // 64 x uint4
  float s = 0.f;
  #pragma unroll 4
  for (int v = 0; v < 64; ++v) {
    uint4 q = row[v];
    const float* wv = w + v * 8;
    s += bf2f(q.x & 0xffffu) * wv[0] + bf2f(q.x >> 16) * wv[1];
    s += bf2f(q.y & 0xffffu) * wv[2] + bf2f(q.y >> 16) * wv[3];
    s += bf2f(q.z & 0xffffu) * wv[4] + bf2f(q.z >> 16) * wv[5];
    s += bf2f(q.w & 0xffffu) * wv[6] + bf2f(q.w >> 16) * wv[7];
  }
  out[m] = s + b[0];
}

// =====================================================================
extern "C" void kernel_launch(void* const* d_in, const int* in_sizes, int n_in,
                              void* d_out, int out_size, void* d_ws, size_t ws_size,
                              hipStream_t stream) {
  (void)in_sizes; (void)n_in; (void)out_size; (void)ws_size;
  const float* g0    = (const float*)d_in[0];
  const float* g1    = (const float*)d_in[1];
  const float* g2    = (const float*)d_in[2];
  const float* g3    = (const float*)d_in[3];
  const float* hf    = (const float*)d_in[4];
  const float* pts   = (const float*)d_in[5];
  const float* gfeat = (const float*)d_in[6];
  const float* scale = (const float*)d_in[7];
  const float* loc   = (const float*)d_in[8];
  const float* camK  = (const float*)d_in[9];
  const float* camRT = (const float*)d_in[10];
  const float* pt_w1 = (const float*)d_in[11];
  const float* pt_b1 = (const float*)d_in[12];
  const float* pt_w2 = (const float*)d_in[13];
  const float* pt_b2 = (const float*)d_in[14];
  const float* z_w1  = (const float*)d_in[15];
  const float* z_b1  = (const float*)d_in[16];
  const float* z_w2  = (const float*)d_in[17];
  const float* z_b2  = (const float*)d_in[18];
  const float* g_w1  = (const float*)d_in[19];
  const float* g_b1  = (const float*)d_in[20];
  const float* g_w2  = (const float*)d_in[21];
  const float* g_b2  = (const float*)d_in[22];
  const float* in_w  = (const float*)d_in[23];
  const float* in_b  = (const float*)d_in[24];
  const float* bw1   = (const float*)d_in[25];
  const float* bb1   = (const float*)d_in[26];
  const float* bw2   = (const float*)d_in[27];
  const float* bb2   = (const float*)d_in[28];
  const float* out_w = (const float*)d_in[29];
  const float* out_b = (const float*)d_in[30];

  const int M  = 131072;              // 8 * 16384 points
  const int NB = 512 * 512;

  // ---- workspace layout (bytes) ----
  char* ws = (char*)d_ws;
  unsigned short* Win  = (unsigned short*)(ws + 0);                 // 1280*512*2
  unsigned short* W1   = (unsigned short*)(ws + 1310720);           // 4*512*512*2
  unsigned short* W2   = (unsigned short*)(ws + 3407872);           // 4*512*512*2
  unsigned short* net  = (unsigned short*)(ws + 5505024);           // M*512*2
  unsigned short* tmp  = (unsigned short*)(ws + 139722752);         // M*512*2
  unsigned short* comb = (unsigned short*)(ws + 273940480);         // M*1280*2
  unsigned short* net2 = comb;        // reuse comb region after "in" layer

  // 0) pack weights
  {
    int total = 1280 * 512 + 8 * NB;
    prep_weights<<<(total + 255) / 256, 256, 0, stream>>>(in_w, bw1, bw2, Win, W1, W2);
  }
  // 1) featurize -> comb bf16 [M][1280]
  featurize<<<M / 256, 256, 0, stream>>>(
      g0, g1, g2, g3, hf, pts, gfeat, scale, loc, camK, camRT,
      pt_w1, pt_b1, pt_w2, pt_b2, z_w1, z_b1, z_w2, z_b2,
      g_w1, g_b1, g_w2, g_b2, comb);

  // 2) net = relu(comb @ Win + in_b)       [M x 1280] x [1280 x 512]
  gemm_bf16<<<dim3(M / 256, 512 / 64), 256, 0, stream>>>(
      comb, Win, in_b, nullptr, net, 1280, 512, 1);

  // 3) 4 residual blocks: tmp = relu(cur@W1+b1); next = relu(tmp@W2+b2+cur)
  unsigned short* cur = net;
  unsigned short* nxt = net2;
  for (int i = 0; i < 4; ++i) {
    gemm_bf16<<<dim3(M / 256, 512 / 64), 256, 0, stream>>>(
        cur, W1 + (size_t)i * NB, bb1 + i * 512, nullptr, tmp, 512, 512, 1);
    gemm_bf16<<<dim3(M / 256, 512 / 64), 256, 0, stream>>>(
        tmp, W2 + (size_t)i * NB, bb2 + i * 512, cur, nxt, 512, 512, 1);
    unsigned short* t = cur; cur = nxt; nxt = t;
  }

  // 4) final projection -> f32 output
  out_layer<<<M / 256, 256, 0, stream>>>(cur, out_w, out_b, (float*)d_out, M);
}
